// StackingGRUNet_45741401703114
// MI455X (gfx1250) — compile-verified, matrix operand order checked by reading, not run on hardware
//
#include <hip/hip_runtime.h>
#include <hip/hip_bf16.h>

#define T_  1024
#define S_  40
#define IN_ 32
#define H_  384
#define G_  1152   // 3*H
#define FC_ 256
#define NC_ 4

// Recurrence decomposition: WPB workgroups per bin, each owns a KS-wide k-slice.
#define WPB 6
#define KS  (H_ / WPB)          // 64
#define KC  (KS / 32)           // 2 WMMA k-chunks
#define RTW (G_ / 16 / 12)      // 6 row-tiles per wave (12 waves/WG)

// CDNA5 async load-to-LDS path (guarded: falls back to sync copies if the
// toolchain doesn't declare these builtins, or on the host pass).
#if defined(__gfx1250__) &&                                              \
    __has_builtin(__builtin_amdgcn_global_load_async_to_lds_b128) &&     \
    __has_builtin(__builtin_amdgcn_s_wait_asynccnt)
#define HAVE_ASYNC_LDS 1
typedef int v4i_t __attribute__((vector_size(16)));
typedef __attribute__((address_space(1))) v4i_t g_v4i;   // global (AS1)
typedef __attribute__((address_space(3))) v4i_t l_v4i;   // LDS    (AS3)
#else
#define HAVE_ASYNC_LDS 0
#endif

typedef __attribute__((ext_vector_type(16))) __bf16          v16bf;
typedef __attribute__((ext_vector_type(8)))  float           v8f;

union FragBF { v16bf v; unsigned u[8]; };

__device__ __forceinline__ float bf2f(unsigned short s) {
  return __uint_as_float(((unsigned)s) << 16);
}
__device__ __forceinline__ unsigned short f2bf(float f) {
  unsigned u = __float_as_uint(f);
  unsigned r = u + 0x7FFFu + ((u >> 16) & 1u);   // round-to-nearest-even
  return (unsigned short)(r >> 16);
}
__device__ __forceinline__ float sigmoidf_(float x) {
  return 1.0f / (1.0f + __expf(-x));
}
__device__ __forceinline__ v8f v8f_zero() {
  v8f z;
#pragma unroll
  for (int i = 0; i < 8; ++i) z[i] = 0.0f;
  return z;
}

// ---------------------------------------------------------------- utility
__global__ void cvt_f32_bf16(const float* __restrict__ in,
                             unsigned short* __restrict__ out, int n) {
  int i = blockIdx.x * blockDim.x + threadIdx.x;
  if (i < n) out[i] = f2bf(in[i]);
}

__global__ void zero_u32(unsigned* __restrict__ p, int n) {
  int i = blockIdx.x * blockDim.x + threadIdx.x;
  if (i < n) p[i] = 0u;
}

// ---------------------------------------------------------------- WMMA GEMM
// Per bin s:  C[m, n] = sum_k A[m, k] * B[n, k]   (k-contiguous operands)
// Each wave computes a 16(M) x 64(N) tile; A fragment reused across 4 WMMAs.
__global__ __launch_bounds__(128) void wmma_gemm(
    const unsigned short* __restrict__ A,
    const unsigned short* __restrict__ B,
    float* __restrict__ C,
    int K, int N,
    long lda, long ldb, long sA, long sB, long sC,
    int tilesM, int tilesN4)
{
  int wave = (int)((blockIdx.x * (unsigned)blockDim.x + threadIdx.x) >> 5);
  int lane = threadIdx.x & 31;
  int perS = tilesM * tilesN4;
  int s   = wave / perS;
  int rem = wave - s * perS;
  int mt  = rem / tilesN4;
  int ng  = rem - mt * tilesN4;
  int m0 = mt << 4;
  int n0 = ng << 6;
  int hi = lane >> 4, lm = lane & 15;

  const unsigned short* Ab  = A + (size_t)s * sA + (size_t)(m0 + lm) * lda;
  const unsigned short* Bb0 = B + (size_t)s * sB;

  v8f acc[4] = {v8f_zero(), v8f_zero(), v8f_zero(), v8f_zero()};

  for (int k0 = 0; k0 < K; k0 += 32) {
    FragBF a;
#pragma unroll
    for (int j = 0; j < 8; ++j) {
      int k = k0 + ((j >> 2) << 4) + (hi << 3) + ((j & 3) << 1);
      a.u[j] = *(const unsigned*)(Ab + k);
    }
#pragma unroll
    for (int nt = 0; nt < 4; ++nt) {
      const unsigned short* Bb = Bb0 + (size_t)(n0 + (nt << 4) + lm) * ldb;
      FragBF b;
#pragma unroll
      for (int j = 0; j < 8; ++j) {
        int k = k0 + ((j >> 2) << 4) + (hi << 3) + ((j & 3) << 1);
        b.u[j] = *(const unsigned*)(Bb + k);
      }
      acc[nt] = __builtin_amdgcn_wmma_f32_16x16x32_bf16(
          false, a.v, false, b.v, (short)0, acc[nt], false, false);
    }
  }

  float* Cs = C + (size_t)s * sC;
#pragma unroll
  for (int nt = 0; nt < 4; ++nt) {
#pragma unroll
    for (int r = 0; r < 8; ++r) {
      int m = m0 + (hi << 3) + r;
      int n = n0 + (nt << 4) + lm;
      Cs[(size_t)m * N + n] = acc[nt][r];
    }
  }
}

// ---------------------------------------------------------------- GRU recurrence
// 6 WGs per bin, k-sliced. W_hh slice (1152x64 bf16 = 147KB) is DMA'd into LDS
// once (async-to-LDS) and stays resident for the whole 1024-step loop. Per
// step each WG computes all 1152 partial gate sums over its k-slice with WMMA
// (h replicated into all 16 B columns), swaps partials through L2 (double-
// buffered), then 32 threads do the gate math for the 64 hidden units this WG
// owns. Each WG's k-slice of h is exactly the slice it updates, so no h
// broadcast is needed. gi for step t+1 is async-prefetched into LDS during
// step t's WMMA phase to hide global latency on the serial critical path.
__global__ __launch_bounds__(384, 1) void gru_mw(
    const unsigned short* __restrict__ Whh,  // [S][G][H] bf16
    const float* __restrict__ bih,           // [S][G]
    const float* __restrict__ bhh,           // [S][G]
    const float* __restrict__ gi,            // [S][T][G] fp32
    float* __restrict__ part,                // [S][WPB][2][G] fp32
    unsigned* __restrict__ barcnt,           // [S] monotonic counters (zeroed)
    unsigned short* __restrict__ hseq,       // [S][T][H] bf16 or nullptr
    float* __restrict__ hlast)               // [S][H] fp32 or nullptr
{
  __shared__ unsigned short sW[G_ * KS];     // 147456 B
  __shared__ unsigned       sHb[KS / 2];     // packed bf16 h pairs
  __shared__ float          sH[KS];          // fp32 h slice
#if HAVE_ASYNC_LDS
  __shared__ float          sGi[2][3 * KS];  // double-buffered gi r/z/n slices
#endif

  const int s   = blockIdx.x / WPB;
  const int w   = blockIdx.x % WPB;
  const int tid = threadIdx.x;
  const int lane = tid & 31, wv = tid >> 5;
  const int lm = lane & 15, hi = lane >> 4;

  const float* gs = gi + (size_t)s * T_ * G_;

  // ---- one-time preload of the W_hh k-slice into LDS (columns w*KS..+KS-1)
#if HAVE_ASYNC_LDS
  for (int r = tid; r < G_; r += 384) {
    const unsigned short* src = Whh + ((size_t)s * G_ + r) * H_ + w * KS;
    unsigned short* dst = sW + (size_t)r * KS;
#pragma unroll
    for (int q = 0; q < KS / 8; ++q)
      __builtin_amdgcn_global_load_async_to_lds_b128(
          (g_v4i*)(src + 8 * q), (l_v4i*)(dst + 8 * q), 0, 0);
  }
  // initial gi prefetch for t = 0 (24 lanes x 32B = 3 segments x 256B)
  if (tid < 24) {
    int seg  = tid >> 3;
    int col8 = (tid & 7) * 8;
    const float* src = gs + (size_t)seg * H_ + w * KS + col8;
    float* dst = &sGi[0][seg * KS + col8];
    __builtin_amdgcn_global_load_async_to_lds_b128(
        (g_v4i*)src, (l_v4i*)dst, 0, 0);
    __builtin_amdgcn_global_load_async_to_lds_b128(
        (g_v4i*)(src + 4), (l_v4i*)(dst + 4), 0, 0);
  }
  __builtin_amdgcn_s_wait_asynccnt(0);
#else
  for (int r = tid; r < G_; r += 384) {
    const uint4* src = (const uint4*)(Whh + ((size_t)s * G_ + r) * H_ + w * KS);
    uint4* dst = (uint4*)(sW + (size_t)r * KS);
#pragma unroll
    for (int q = 0; q < KS / 8; ++q) dst[q] = src[q];
  }
#endif
  if (tid < KS)     sH[tid] = 0.0f;
  if (tid < KS / 2) sHb[tid] = 0u;
  __syncthreads();

  // ---- per-thread gate biases (only gate-math threads use these)
  float bri[2], brh[2], bzi[2], bzh[2], bni[2], bnh[2];
  if (tid < KS / 2) {
#pragma unroll
    for (int e = 0; e < 2; ++e) {
      int jh = w * KS + 2 * tid + e;
      bri[e] = bih[s * G_ + jh];            brh[e] = bhh[s * G_ + jh];
      bzi[e] = bih[s * G_ + H_ + jh];       bzh[e] = bhh[s * G_ + H_ + jh];
      bni[e] = bih[s * G_ + 2 * H_ + jh];   bnh[e] = bhh[s * G_ + 2 * H_ + jh];
    }
  }

  const unsigned* sWu = (const unsigned*)sW;
  float* partMine = part + ((size_t)s * WPB + w) * 2 * G_;

  for (int t = 0; t < T_; ++t) {
    const int buf = t & 1;

#if HAVE_ASYNC_LDS
    // ---------- async prefetch of gi for the next step (always 2 transfers
    // in flight per wave-0 lane; last iteration targets the dead buffer)
    if (tid < 24) {
      const int tp  = (t + 1 < T_) ? (t + 1) : 0;
      const int pb  = (t + 1) & 1;
      int seg  = tid >> 3;
      int col8 = (tid & 7) * 8;
      const float* src = gs + (size_t)tp * G_ + seg * H_ + w * KS + col8;
      float* dst = &sGi[pb][seg * KS + col8];
      __builtin_amdgcn_global_load_async_to_lds_b128(
          (g_v4i*)src, (l_v4i*)dst, 0, 0);
      __builtin_amdgcn_global_load_async_to_lds_b128(
          (g_v4i*)(src + 4), (l_v4i*)(dst + 4), 0, 0);
    }
#endif

    // ---------- phase A: WMMA partial sums (all 384 threads, EXEC all-ones)
    FragBF bfr[KC];   // h fragments, identical in all 16 B columns
#pragma unroll
    for (int c = 0; c < KC; ++c) {
      int base2 = ((c << 5) + (hi << 3)) >> 1;
#pragma unroll
      for (int j = 0; j < 8; ++j)
        bfr[c].u[j] = sHb[base2 + (j & 3) + ((j >> 2) << 3)];
    }
    float* pout = partMine + (size_t)buf * G_;
#pragma unroll
    for (int i = 0; i < RTW; ++i) {
      int rt = wv * RTW + i;
      v8f acc = v8f_zero();
#pragma unroll
      for (int c = 0; c < KC; ++c) {
        FragBF a;
        int base = ((rt * 16 + lm) * KS + (c << 5) + (hi << 3)) >> 1;
#pragma unroll
        for (int j = 0; j < 8; ++j)
          a.u[j] = sWu[base + (j & 3) + ((j >> 2) << 3)];
        acc = __builtin_amdgcn_wmma_f32_16x16x32_bf16(
            false, a.v, false, bfr[c].v, (short)0, acc, false, false);
      }
      if (lm == 0) {   // all D columns identical; column-0 lanes publish rows
        float* pp = pout + rt * 16 + (hi << 3);
#pragma unroll
        for (int r = 0; r < 8; ++r) pp[r] = acc[r];
      }
    }

    // ---------- inter-WG barrier (per bin, monotonic counter)
    if (tid == 0) {
      __threadfence();
      __hip_atomic_fetch_add(&barcnt[s], 1u, __ATOMIC_RELEASE,
                             __HIP_MEMORY_SCOPE_AGENT);
      const unsigned target = (unsigned)WPB * (unsigned)(t + 1);
      while (__hip_atomic_load(&barcnt[s], __ATOMIC_ACQUIRE,
                               __HIP_MEMORY_SCOPE_AGENT) < target)
        __builtin_amdgcn_s_sleep(1);
    }
    __syncthreads();

    // ---------- phase B: gate math for the KS units this WG owns
    if (tid < KS / 2) {
#if HAVE_ASYNC_LDS
      // retire step-t's gi transfers; step-(t+1)'s 2 may stay in flight
      __builtin_amdgcn_s_wait_asynccnt(2);
#endif
      const float* g = gs + (size_t)t * G_;
      (void)g;
      float hn2[2];
#pragma unroll
      for (int e = 0; e < 2; ++e) {
        int u  = 2 * tid + e;
        int jh = w * KS + u;
        float sr = 0.0f, sz = 0.0f, sn = 0.0f;
#pragma unroll
        for (int q = 0; q < WPB; ++q) {
          const float* pq = part + (((size_t)s * WPB + q) * 2 + buf) * G_;
          sr += pq[jh];
          sz += pq[H_ + jh];
          sn += pq[2 * H_ + jh];
        }
#if HAVE_ASYNC_LDS
        float gr = sGi[buf][u];
        float gz = sGi[buf][KS + u];
        float gn = sGi[buf][2 * KS + u];
#else
        float gr = g[jh];
        float gz = g[H_ + jh];
        float gn = g[2 * H_ + jh];
#endif
        float r = sigmoidf_(gr + bri[e] + sr + brh[e]);
        float z = sigmoidf_(gz + bzi[e] + sz + bzh[e]);
        float n = tanhf    (gn + bni[e] + r * (sn + bnh[e]));
        hn2[e] = (1.0f - z) * n + z * sH[u];
      }
      sH[2 * tid]     = hn2[0];
      sH[2 * tid + 1] = hn2[1];
      unsigned pk = (unsigned)f2bf(hn2[0]) | ((unsigned)f2bf(hn2[1]) << 16);
      sHb[tid] = pk;
      if (hseq) {
        unsigned* hp = (unsigned*)hseq;
        hp[(((size_t)s * T_ + t) * H_ + w * KS) / 2 + tid] = pk;
      }
    }
    __syncthreads();
  }

  if (hlast && tid < KS) hlast[(size_t)s * H_ + w * KS + tid] = sH[tid];
}

// ---------------------------------------------------------------- FC + ReLU
__global__ __launch_bounds__(FC_) void fc_relu(
    const float* __restrict__ Wfc,   // [S][FC][H]
    const float* __restrict__ bfc,   // [S][FC]
    const float* __restrict__ last,  // [S][H]
    float* __restrict__ feat)        // [S][FC]
{
  int s = blockIdx.x, f = threadIdx.x;
  __shared__ float l[H_];
  for (int k = threadIdx.x; k < H_; k += FC_) l[k] = last[(size_t)s * H_ + k];
  __syncthreads();
  const float* wp = Wfc + ((size_t)s * FC_ + f) * H_;
  float a = bfc[s * FC_ + f];
  for (int k = 0; k < H_; ++k) a = fmaf(wp[k], l[k], a);
  feat[s * FC_ + f] = a > 0.0f ? a : 0.0f;
}

// ---------------------------------------------------------------- head + log_softmax
__global__ __launch_bounds__(256) void out_head(
    const float* __restrict__ Wout,  // [NC][S*FC]
    const float* __restrict__ bout,  // [NC]
    const float* __restrict__ feat,  // [S*FC] flat
    float* __restrict__ out)         // [NC]
{
  __shared__ float red[NC_][256];
  int j = threadIdx.x;
  for (int c = 0; c < NC_; ++c) {
    float a = 0.0f;
    for (int k = j; k < S_ * FC_; k += 256)
      a = fmaf(Wout[(size_t)c * S_ * FC_ + k], feat[k], a);
    red[c][j] = a;
  }
  __syncthreads();
  for (int st = 128; st > 0; st >>= 1) {
    if (j < st)
      for (int c = 0; c < NC_; ++c) red[c][j] += red[c][j + st];
    __syncthreads();
  }
  if (j == 0) {
    float l[NC_];
    float m = -3.4e38f;
    for (int c = 0; c < NC_; ++c) { l[c] = red[c][0] + bout[c]; m = fmaxf(m, l[c]); }
    float ssum = 0.0f;
    for (int c = 0; c < NC_; ++c) ssum += expf(l[c] - m);
    float lse = m + logf(ssum);
    for (int c = 0; c < NC_; ++c) out[c] = l[c] - lse;
  }
}

// ---------------------------------------------------------------- launch
extern "C" void kernel_launch(void* const* d_in, const int* in_sizes, int n_in,
                              void* d_out, int out_size, void* d_ws, size_t ws_size,
                              hipStream_t stream) {
  (void)in_sizes; (void)n_in; (void)out_size; (void)ws_size;

  const float* x    = (const float*)d_in[0];
  const float* Wih0 = (const float*)d_in[1];
  const float* Whh0 = (const float*)d_in[2];
  const float* bih0 = (const float*)d_in[3];
  const float* bhh0 = (const float*)d_in[4];
  const float* Wih1 = (const float*)d_in[5];
  const float* Whh1 = (const float*)d_in[6];
  const float* bih1 = (const float*)d_in[7];
  const float* bhh1 = (const float*)d_in[8];
  const float* Wfc  = (const float*)d_in[9];
  const float* bfc  = (const float*)d_in[10];
  const float* Wout = (const float*)d_in[11];
  const float* bout = (const float*)d_in[12];
  float* out = (float*)d_out;

  char* ws = (char*)d_ws;
  size_t off = 0;
  auto alloc = [&](size_t bytes) -> void* {
    void* p = ws + off;
    off += (bytes + 255) & ~(size_t)255;
    return p;
  };

  unsigned short* xb     = (unsigned short*)alloc((size_t)T_ * S_ * IN_ * 2);
  unsigned short* Wih0b  = (unsigned short*)alloc((size_t)S_ * G_ * IN_ * 2);
  unsigned short* Whh0b  = (unsigned short*)alloc((size_t)S_ * G_ * H_ * 2);
  unsigned short* Wih1b  = (unsigned short*)alloc((size_t)S_ * G_ * H_ * 2);
  unsigned short* Whh1b  = (unsigned short*)alloc((size_t)S_ * G_ * H_ * 2);
  float*          gi0    = (float*)alloc((size_t)S_ * T_ * G_ * 4);
  float*          gi1    = (float*)alloc((size_t)S_ * T_ * G_ * 4);
  unsigned short* h1b    = (unsigned short*)alloc((size_t)S_ * T_ * H_ * 2);
  float*          h2last = (float*)alloc((size_t)S_ * H_ * 4);
  float*          feat   = (float*)alloc((size_t)S_ * FC_ * 4);
  float*          part   = (float*)alloc((size_t)S_ * WPB * 2 * G_ * 4);
  unsigned*       barcnt = (unsigned*)alloc((size_t)2 * S_ * 4);

  zero_u32<<<1, 128, 0, stream>>>(barcnt, 2 * S_);

  auto cvt = [&](const float* src, unsigned short* dst, int n) {
    cvt_f32_bf16<<<(n + 255) / 256, 256, 0, stream>>>(src, dst, n);
  };
  cvt(x,    xb,    T_ * S_ * IN_);
  cvt(Wih0, Wih0b, S_ * G_ * IN_);
  cvt(Whh0, Whh0b, S_ * G_ * H_);
  cvt(Wih1, Wih1b, S_ * G_ * H_);
  cvt(Whh1, Whh1b, S_ * G_ * H_);

  const int tilesM  = T_ / 16;   // 64
  const int tilesN4 = G_ / 64;   // 18
  const int waves   = S_ * tilesM * tilesN4;       // 46080
  const int blocks  = waves * 32 / 128;            // 11520

  // gi0[s] = x_s[T,IN] * W_ih0[s]^T   (K = 32)
  wmma_gemm<<<blocks, 128, 0, stream>>>(
      xb, Wih0b, gi0, IN_, G_,
      (long)(S_ * IN_), (long)IN_,
      (long)IN_, (long)G_ * IN_, (long)T_ * G_, tilesM, tilesN4);

  // layer-0 recurrence -> h1 sequence (bf16)
  gru_mw<<<S_ * WPB, 384, 0, stream>>>(
      Whh0b, bih0, bhh0, gi0, part, barcnt, h1b, nullptr);

  // gi1[s] = h1_s[T,H] * W_ih1[s]^T   (K = 384; FLOP-dominant GEMM)
  wmma_gemm<<<blocks, 128, 0, stream>>>(
      h1b, Wih1b, gi1, H_, G_,
      (long)H_, (long)H_,
      (long)T_ * H_, (long)G_ * H_, (long)T_ * G_, tilesM, tilesN4);

  // layer-1 recurrence -> only final hidden state
  gru_mw<<<S_ * WPB, 384, 0, stream>>>(
      Whh1b, bih1, bhh1, gi1, part, barcnt + S_, nullptr, h2last);

  fc_relu<<<S_, FC_, 0, stream>>>(Wfc, bfc, h2last, feat);
  out_head<<<1, 256, 0, stream>>>(Wout, bout, feat, out);
}
